// AttnDecoderRNN_28707561407114
// MI455X (gfx1250) — matrix operand order repaired and checked by reading, not compiled, not run on hardware
//
#include <hip/hip_runtime.h>
#include <hip/hip_bf16.h>

// ---------------------------------------------------------------------------
// CDNA5 (gfx1250) wave32 WMMA bf16 types
// ---------------------------------------------------------------------------
typedef __bf16 bf16_t;
typedef __attribute__((ext_vector_type(16))) __bf16 v16bf;
typedef __attribute__((ext_vector_type(8)))  __bf16 v8bf;
typedef __attribute__((ext_vector_type(8)))  float  v8f;

#define B_SZ   64
#define EMB_SZ 1024
#define HD_SZ  1024
#define L_SZ   128
#define V_SZ   50257

// f32 -> bf16: plain cast lowers to native v_cvt_pk_bf16_f32 on gfx1250
__device__ __forceinline__ bf16_t f2bf(float f) { return (bf16_t)f; }

// ---------------------------------------------------------------------------
// Fragment loaders (wave32): contiguous k order (k = 16h + s), same in A and B
// A 16x32 (MxK): lanes 0-15 -> M=lane, K=0..15;
//                lanes 16-31 -> M=lane-16, K=16..31
// A is pre-converted bf16: two 16B vector loads, no VALU conversion.
// ---------------------------------------------------------------------------
__device__ __forceinline__ v16bf load_a_frag(const bf16_t* __restrict__ A0,
                                             const bf16_t* __restrict__ A1,
                                             int K0, int K1,
                                             int m0, int k0, int lane) {
    int m = m0 + (lane & 15);
    const bf16_t* src;
    if (k0 < K0) src = A0 + (size_t)m * K0 + k0;
    else         src = A1 + (size_t)m * K1 + (k0 - K0);
    int off = (lane >> 4) << 4;                    // +16 elems for upper half-lanes
    v8bf lo = *(const v8bf*)(src + off);           // K = base .. base+7
    v8bf hi = *(const v8bf*)(src + off + 8);       // K = base+8 .. base+15
    return __builtin_shufflevector(lo, hi, 0,1,2,3,4,5,6,7,8,9,10,11,12,13,14,15);
}

// B 32x16 (KxN) from W[N,K] f32 row-major (B[k][n] = W[n][k]):
// lanes 0-15 -> N=lane, K=0..15 ; lanes 16-31 -> N=lane-16, K=16..31
__device__ __forceinline__ v16bf load_b_frag(const float* __restrict__ W, int Kdim,
                                             int n0, int k0, int lane, int Nmax) {
    int n = n0 + (lane & 15);
    if (n >= Nmax) n = Nmax - 1;                   // tail tile clamp (store-masked)
    const float* row = W + (size_t)n * Kdim + k0 + ((lane >> 4) << 4);
    __builtin_prefetch(row + 64, 0, 1);            // -> global_prefetch_b8
    float4 q0 = *(const float4*)(row + 0);
    float4 q1 = *(const float4*)(row + 4);
    float4 q2 = *(const float4*)(row + 8);
    float4 q3 = *(const float4*)(row + 12);
    v16bf b;
    b[0] = f2bf(q0.x); b[1] = f2bf(q0.y); b[2]  = f2bf(q0.z); b[3]  = f2bf(q0.w);
    b[4] = f2bf(q1.x); b[5] = f2bf(q1.y); b[6]  = f2bf(q1.z); b[7]  = f2bf(q1.w);
    b[8] = f2bf(q2.x); b[9] = f2bf(q2.y); b[10] = f2bf(q2.z); b[11] = f2bf(q2.w);
    b[12]= f2bf(q3.x); b[13]= f2bf(q3.y); b[14] = f2bf(q3.z); b[15] = f2bf(q3.w);
    return b;
}

// C/D 16x16 f32: VGPR r -> rows m0+r (lanes 0-15) / m0+8+r (lanes 16-31), N=lane&15
// FLAGS: bit0 = relu, bit1 = store bf16 (compile-time: no epilogue branching)
template <int FLAGS>
__device__ __forceinline__ void store_tile(void* Out, int ldo, int m0, int n0,
                                           int Nmax, const float* bias,
                                           v8f acc, int lane) {
    int col = n0 + (lane & 15);
    if (col >= Nmax) return;
    float bv = bias[col];
    int rbase = m0 + ((lane >> 4) << 3);
#pragma unroll
    for (int r = 0; r < 8; ++r) {
        float v = acc[r] + bv;
        if (FLAGS & 1) v = fmaxf(v, 0.0f);
        size_t idx = (size_t)(rbase + r) * ldo + col;
        if (FLAGS & 2) ((bf16_t*)Out)[idx] = f2bf(v);
        else           ((float*)Out)[idx]  = v;
    }
}

// ---------------------------------------------------------------------------
// Generic M=64 GEMM: Out[64,N] = [A0 | A1][64,Ktot](bf16) @ W[N,Ktot](f32)^T + bias
// One wave per 16-wide N tile; the wave accumulates all four 16-row M tiles so
// each streamed W tile is converted once and used 4x from registers. All five
// fragments of a k-step are materialized before the WMMAs so their loads can
// issue as clauses and overlap.
// ---------------------------------------------------------------------------
template <int FLAGS>
__global__ __launch_bounds__(256) void
wmma_gemm64(const bf16_t* __restrict__ A0, const bf16_t* __restrict__ A1,
            int K0, int Ktot,
            const float* __restrict__ W, const float* __restrict__ bias,
            void* __restrict__ Out, int N) {
    const int lane = threadIdx.x & 31;
    const int wid  = (blockIdx.x * blockDim.x + threadIdx.x) >> 5;
    const int ntiles = (N + 15) >> 4;
    if (wid >= ntiles) return;                 // uniform per-wave: EXEC stays all-1s
    const int n0 = wid << 4;
    const int K1 = Ktot - K0;

    v8f acc[4];
#pragma unroll
    for (int t = 0; t < 4; ++t)
#pragma unroll
        for (int r = 0; r < 8; ++r) acc[t][r] = 0.0f;

#pragma unroll 2
    for (int k0 = 0; k0 < Ktot; k0 += 32) {
        v16bf af0 = load_a_frag(A0, A1, K0, K1, 0,  k0, lane);
        v16bf af1 = load_a_frag(A0, A1, K0, K1, 16, k0, lane);
        v16bf af2 = load_a_frag(A0, A1, K0, K1, 32, k0, lane);
        v16bf af3 = load_a_frag(A0, A1, K0, K1, 48, k0, lane);
        v16bf bf  = load_b_frag(W, Ktot, n0, k0, lane, N);
        acc[0] = __builtin_amdgcn_wmma_f32_16x16x32_bf16(false, af0, false, bf,
                                                         (short)0, acc[0], false, false);
        acc[1] = __builtin_amdgcn_wmma_f32_16x16x32_bf16(false, af1, false, bf,
                                                         (short)0, acc[1], false, false);
        acc[2] = __builtin_amdgcn_wmma_f32_16x16x32_bf16(false, af2, false, bf,
                                                         (short)0, acc[2], false, false);
        acc[3] = __builtin_amdgcn_wmma_f32_16x16x32_bf16(false, af3, false, bf,
                                                         (short)0, acc[3], false, false);
    }
#pragma unroll
    for (int t = 0; t < 4; ++t)
        store_tile<FLAGS>(Out, N, t * 16, n0, N, bias, acc[t], lane);
}

// ---------------------------------------------------------------------------
// Small helpers
// ---------------------------------------------------------------------------
__global__ void embed_kernel(const int* __restrict__ idx,
                             const float* __restrict__ table,
                             bf16_t* __restrict__ out) {
    int b = blockIdx.x;
    const float* row = table + (size_t)idx[b] * EMB_SZ;
    for (int e = threadIdx.x; e < EMB_SZ; e += blockDim.x)
        out[b * EMB_SZ + e] = f2bf(row[e]);
}

__global__ void cvt_bf16_kernel(const float* __restrict__ in,
                                bf16_t* __restrict__ out, int n) {
    int i = blockIdx.x * blockDim.x + threadIdx.x;
    if (i < n) out[i] = f2bf(in[i]);
}

__global__ void softmax128_kernel(const float* __restrict__ in,
                                  float* __restrict__ out) {
    __shared__ float s[L_SZ];
    int b = blockIdx.x, t = threadIdx.x;
    float v = in[b * L_SZ + t];
    s[t] = v; __syncthreads();
    for (int o = 64; o > 0; o >>= 1) { if (t < o) s[t] = fmaxf(s[t], s[t + o]); __syncthreads(); }
    float m = s[0]; __syncthreads();
    float e = expf(v - m);
    s[t] = e; __syncthreads();
    for (int o = 64; o > 0; o >>= 1) { if (t < o) s[t] += s[t + o]; __syncthreads(); }
    out[b * L_SZ + t] = e / s[0];
}

// ctx[b,e] = sum_l attn[b,l] * enc[l,b,e]   (per-b matrix: not a plain GEMM)
__global__ void ctx_kernel(const float* __restrict__ enc,
                           const float* __restrict__ attn,
                           bf16_t* __restrict__ ctx_bf) {
    __shared__ float w[L_SZ];
    int b = blockIdx.x;
    int e = blockIdx.y * blockDim.x + threadIdx.x;
    w[threadIdx.x] = attn[b * L_SZ + threadIdx.x];
    __syncthreads();
    float acc = 0.0f;
#pragma unroll 4
    for (int l = 0; l < L_SZ; ++l)
        acc += w[l] * enc[((size_t)l * B_SZ + b) * HD_SZ + e];
    ctx_bf[b * HD_SZ + e] = f2bf(acc);
}

__global__ void gru_gate_kernel(const float* __restrict__ gi,
                                const float* __restrict__ gh,
                                const float* __restrict__ h0,
                                float* __restrict__ hnew,
                                bf16_t* __restrict__ hnew_bf) {
    int i = blockIdx.x * blockDim.x + threadIdx.x;   // 64*1024
    int b = i >> 10, j = i & 1023;
    const float* gib = gi + b * 3 * HD_SZ;
    const float* ghb = gh + b * 3 * HD_SZ;
    float r = 1.0f / (1.0f + expf(-(gib[j] + ghb[j])));
    float z = 1.0f / (1.0f + expf(-(gib[HD_SZ + j] + ghb[HD_SZ + j])));
    float n = tanhf(gib[2 * HD_SZ + j] + r * ghb[2 * HD_SZ + j]);
    float h = (1.0f - z) * n + z * h0[i];
    hnew[i] = h;
    hnew_bf[i] = f2bf(h);
}

__global__ void lse_stats_kernel(const float* __restrict__ logits,
                                 float* __restrict__ stats) {
    __shared__ float s[256];
    int b = blockIdx.x, t = threadIdx.x;
    const float* row = logits + (size_t)b * V_SZ;
    float m = -3.402823466e38f;
    for (int i = t; i < V_SZ; i += 256) m = fmaxf(m, row[i]);
    s[t] = m; __syncthreads();
    for (int o = 128; o > 0; o >>= 1) { if (t < o) s[t] = fmaxf(s[t], s[t + o]); __syncthreads(); }
    m = s[0]; __syncthreads();
    float sum = 0.0f;
    for (int i = t; i < V_SZ; i += 256) sum += expf(row[i] - m);
    s[t] = sum; __syncthreads();
    for (int o = 128; o > 0; o >>= 1) { if (t < o) s[t] += s[t + o]; __syncthreads(); }
    if (t == 0) { stats[2 * b] = m; stats[2 * b + 1] = logf(s[0]); }
}

__global__ void lse_apply_kernel(float* __restrict__ logits,
                                 const float* __restrict__ stats, int n) {
    int i = blockIdx.x * blockDim.x + threadIdx.x;
    if (i >= n) return;
    int b = i / V_SZ;
    logits[i] = logits[i] - stats[2 * b] - stats[2 * b + 1];
}

// ---------------------------------------------------------------------------
// Launch
// ---------------------------------------------------------------------------
extern "C" void kernel_launch(void* const* d_in, const int* in_sizes, int n_in,
                              void* d_out, int out_size, void* d_ws, size_t ws_size,
                              hipStream_t stream) {
    (void)in_sizes; (void)n_in; (void)out_size; (void)ws_size;
    const int*   input_t = (const int*)  d_in[0];
    const float* hidden  = (const float*)d_in[1];   // [64,1024]
    const float* enc     = (const float*)d_in[2];   // [128,64,1024]
    const float* emb_tab = (const float*)d_in[4];
    const float* attn_w  = (const float*)d_in[5];
    const float* attn_b  = (const float*)d_in[6];
    const float* comb_w  = (const float*)d_in[7];
    const float* comb_b  = (const float*)d_in[8];
    const float* w_ih    = (const float*)d_in[9];
    const float* w_hh    = (const float*)d_in[10];
    const float* b_ih    = (const float*)d_in[11];
    const float* b_hh    = (const float*)d_in[12];
    const float* out_w   = (const float*)d_in[13];
    const float* out_b   = (const float*)d_in[14];

    // workspace layout (byte offsets, all 16B aligned)
    char* ws = (char*)d_ws;
    bf16_t* embedded_bf = (bf16_t*)(ws + 0);         // 64*1024 bf16 = 128KB
    bf16_t* hidden_bf   = (bf16_t*)(ws + 131072);    // 128KB
    bf16_t* ctx_bf      = (bf16_t*)(ws + 262144);    // 128KB
    bf16_t* x_bf        = (bf16_t*)(ws + 393216);    // 128KB
    bf16_t* hnew_bf     = (bf16_t*)(ws + 524288);    // 128KB
    float*  attn_log    = (float*) (ws + 655360);    // 64*128 f32 = 32KB
    float*  gi          = (float*) (ws + 688128);    // 64*3072 f32 = 768KB
    float*  gh          = (float*) (ws + 1474560);   // 768KB
    float*  stats       = (float*) (ws + 2260992);   // 512B

    float* logits = (float*)d_out;                       // [64,50257]
    float* h_new  = logits + (size_t)B_SZ * V_SZ;        // [64,1024]
    float* attn_o = h_new + B_SZ * HD_SZ;                // [64,128]

    // 1) embedding gather (-> bf16) and hidden convert (-> bf16)
    embed_kernel<<<B_SZ, 256, 0, stream>>>(input_t, emb_tab, embedded_bf);
    cvt_bf16_kernel<<<(B_SZ * HD_SZ) / 256, 256, 0, stream>>>(hidden, hidden_bf,
                                                              B_SZ * HD_SZ);
    // 2) attention logits: [emb|h0] @ attn_w^T  (N=128 -> 8 N-tiles -> 1 block)
    wmma_gemm64<0><<<1, 256, 0, stream>>>(embedded_bf, hidden_bf, 1024, 2048,
                                          attn_w, attn_b, attn_log, L_SZ);
    // 3) softmax over L
    softmax128_kernel<<<B_SZ, L_SZ, 0, stream>>>(attn_log, attn_o);
    // 4) context reduction (-> bf16)
    ctx_kernel<<<dim3(B_SZ, HD_SZ / 128), 128, 0, stream>>>(enc, attn_o, ctx_bf);
    // 5) combine + ReLU: [emb|ctx] @ comb_w^T -> x (bf16 store; only feeds next GEMM)
    wmma_gemm64<3><<<8, 256, 0, stream>>>(embedded_bf, ctx_bf, 1024, 2048,
                                          comb_w, comb_b, x_bf, HD_SZ);
    // 6) GRU input/hidden GEMMs (N=3072 -> 192 N-tiles -> 24 blocks each)
    wmma_gemm64<0><<<24, 256, 0, stream>>>(x_bf, x_bf, 1024, 1024,
                                           w_ih, b_ih, gi, 3 * HD_SZ);
    wmma_gemm64<0><<<24, 256, 0, stream>>>(hidden_bf, hidden_bf, 1024, 1024,
                                           w_hh, b_hh, gh, 3 * HD_SZ);
    // 7) gates -> h_new (f32 output) + bf16 copy for the projection
    gru_gate_kernel<<<(B_SZ * HD_SZ) / 256, 256, 0, stream>>>(gi, gh, hidden,
                                                              h_new, hnew_bf);
    // 8) vocab projection (dominant: 206 MB weight stream; 3142 N-tiles -> 393 blocks)
    wmma_gemm64<0><<<393, 256, 0, stream>>>(hnew_bf, hnew_bf, 1024, 1024,
                                            out_w, out_b, logits, V_SZ);
    // 9) log_softmax (two-pass)
    lse_stats_kernel<<<B_SZ, 256, 0, stream>>>(logits, stats);
    int n = B_SZ * V_SZ;
    lse_apply_kernel<<<(n + 255) / 256, 256, 0, stream>>>(logits, stats, n);
}
